// GaussMemoryStep_11373073400445
// MI455X (gfx1250) — matrix operand order repaired and checked by reading, not compiled, not run on hardware
//
#include <hip/hip_runtime.h>
#include <hip/hip_bf16.h>

// ---------------------------------------------------------------------------
// GaussMemory on MI455X (gfx1250): everything is GEMM.
//   rfft-project  -> x @ (F @ w^T)       (F generated analytically, bf16)
//   synthesize    -> retrieved @ (F @ wo)^T * (out_scale * 2/V)
// All heavy math through v_wmma_f32_16x16x32_bf16 (wave32 WMMA).
// ---------------------------------------------------------------------------

typedef __bf16 bf16;
typedef __attribute__((ext_vector_type(16))) __bf16 v16bf;
typedef __attribute__((ext_vector_type(8)))  __bf16 v8bf;
typedef __attribute__((ext_vector_type(8)))  float  v8f;

#define BM 128
#define BN 128
#define BK 32
#define LDSA 40   // 32 + 8 pad (80B pitch -> conflict-free frag loads)
#define LDSB 40

// F[v,j]: j<256 -> cos(2*pi*(j+1)*v/8192) ; j>=256 -> -sin(2*pi*(j-255)*v/8192)
__global__ void gen_fourier_bf16(bf16* __restrict__ F, int total) {
  int idx = blockIdx.x * blockDim.x + threadIdx.x;
  if (idx >= total) return;
  int v = idx >> 9;          // 512 cols
  int j = idx & 511;
  int f = (j & 255) + 1;
  int r = (int)(((long long)f * (long long)v) & 8191);   // exact phase mod V
  float theta = (float)r * 7.6699039394282066e-4f;       // 2*pi/8192
  float val = (j < 256) ? __cosf(theta) : -__sinf(theta);
  F[idx] = (bf16)val;
}

__global__ void cvt_f32_bf16(const float* __restrict__ src, bf16* __restrict__ dst, int n) {
  int i = blockIdx.x * blockDim.x + threadIdx.x;
  if (i < n) dst[i] = (bf16)src[i];
}

// Generic tiled WMMA GEMM: C = A(MxK) @ B + epilogue
//   A_F32  : A is f32 in global, converted to bf16 on the LDS-fill path
//   TRANS_B: B stored row-major [N,K] (i.e. B^T); else row-major [K,N]
//   EPI    : 0 none -> bf16 C ; 1 causal-decay mask (scores) ; 2 scale -> f32 C
// Requires M%128==0, N%128==0, K%32==0 (true for all shapes here).
template <bool A_F32, bool TRANS_B, int EPI, typename OutT>
__global__ __launch_bounds__(256) void wmma_gemm(
    const void* __restrict__ Aptr, const bf16* __restrict__ Bptr,
    OutT* __restrict__ Cptr, int M, int N, int K, int lda, int ldb, int ldc,
    long long strideA, long long strideB, long long strideC,
    const float* __restrict__ decay_logit, const float* __restrict__ out_scale) {
  (void)M;
  __shared__ bf16 sA[BM * LDSA];
  __shared__ bf16 sB[BN * LDSB];

  const int tid  = threadIdx.x;
  const int lane = tid & 31;
  const int wave = tid >> 5;
  const int l15  = lane & 15;
  const bool hi  = lane >= 16;

  const int bm = blockIdx.y * BM;
  const int bn = blockIdx.x * BN;
  const long long zb = blockIdx.z;

  const float* Af = (const float*)Aptr + zb * strideA;
  const bf16*  Ab = (const bf16*)Aptr + zb * strideA;
  const bf16*  B  = Bptr + zb * strideB;
  OutT*        Cc = Cptr + zb * strideC;

  // stage-in index maps (256 threads)
  const int arow = tid >> 1;           // 0..127
  const int acol = (tid & 1) * 16;     // 0 / 16
  const int bk   = tid >> 3;           // 0..31   (TRANS_B == 0)
  const int bnn  = (tid & 7) * 16;     // 0..112

  const int wm = (wave & 3) * 32;      // 4 waves along M
  const int wn = (wave >> 2) * 64;     // 2 waves along N

  v8f acc[2][4];
#pragma unroll
  for (int i = 0; i < 2; ++i)
#pragma unroll
    for (int j = 0; j < 4; ++j)
#pragma unroll
      for (int r = 0; r < 8; ++r) acc[i][j][r] = 0.0f;

  for (int k0 = 0; k0 < K; k0 += BK) {
    // ---- A tile -> LDS (row-major, K contiguous) ----
    if (A_F32) {
      const float* p = Af + (size_t)(bm + arow) * lda + k0 + acol;
      float tf[16];
      *(float4*)&tf[0]  = *(const float4*)(p);
      *(float4*)&tf[4]  = *(const float4*)(p + 4);
      *(float4*)&tf[8]  = *(const float4*)(p + 8);
      *(float4*)&tf[12] = *(const float4*)(p + 12);
      bf16 t[16];
#pragma unroll
      for (int i = 0; i < 16; ++i) t[i] = (bf16)tf[i];
      *(v8bf*)&sA[arow * LDSA + acol]     = *(v8bf*)&t[0];
      *(v8bf*)&sA[arow * LDSA + acol + 8] = *(v8bf*)&t[8];
    } else {
      const bf16* p = Ab + (size_t)(bm + arow) * lda + k0 + acol;
      *(v8bf*)&sA[arow * LDSA + acol]     = *(const v8bf*)(p);
      *(v8bf*)&sA[arow * LDSA + acol + 8] = *(const v8bf*)(p + 8);
    }
    // ---- B tile -> LDS as [n][k] (K contiguous) ----
    if (TRANS_B) {
      const bf16* p = B + (size_t)(bn + arow) * ldb + k0 + acol;
      *(v8bf*)&sB[arow * LDSB + acol]     = *(const v8bf*)(p);
      *(v8bf*)&sB[arow * LDSB + acol + 8] = *(const v8bf*)(p + 8);
    } else {
      const bf16* p = B + (size_t)(k0 + bk) * ldb + bn + bnn;
      bf16 t[16];
      *(v8bf*)&t[0] = *(const v8bf*)(p);
      *(v8bf*)&t[8] = *(const v8bf*)(p + 8);
#pragma unroll
      for (int i = 0; i < 16; ++i) sB[(bnn + i) * LDSB + bk] = t[i];
    }
    __syncthreads();

    // ---- WMMA fragments + matmul ----
    v16bf afr[2];
#pragma unroll
    for (int i = 0; i < 2; ++i) {
      const bf16* pa = &sA[(wm + i * 16 + l15) * LDSA + (hi ? 8 : 0)];
      ((v8bf*)&afr[i])[0] = *(const v8bf*)(pa);        // K 0..7  / 8..15
      ((v8bf*)&afr[i])[1] = *(const v8bf*)(pa + 16);   // K 16..23 / 24..31
    }
#pragma unroll
    for (int j = 0; j < 4; ++j) {
      const bf16* pb = &sB[(wn + j * 16 + l15) * LDSB + (hi ? 8 : 0)];
      v16bf bfr;
      ((v8bf*)&bfr)[0] = *(const v8bf*)(pb);
      ((v8bf*)&bfr)[1] = *(const v8bf*)(pb + 16);
#pragma unroll
      for (int i = 0; i < 2; ++i) {
        acc[i][j] = __builtin_amdgcn_wmma_f32_16x16x32_bf16(
            false, afr[i], false, bfr, (short)0, acc[i][j], false, false);
      }
    }
    __syncthreads();
  }

  // ---- epilogue ----
  float ldec = 0.0f, scale = 1.0f;
  if (EPI == 1) {
    float d = 1.0f / (1.0f + __expf(-decay_logit[0]));  // sigmoid
    ldec = __logf(d);
  }
  if (EPI == 2) scale = out_scale[0] * (2.0f / 8192.0f);  // fold irfft 2/V

#pragma unroll
  for (int i = 0; i < 2; ++i)
#pragma unroll
    for (int j = 0; j < 4; ++j) {
      const int row0 = bm + wm + i * 16 + (hi ? 8 : 0);
      const int col  = bn + wn + j * 16 + l15;
#pragma unroll
      for (int r = 0; r < 8; ++r) {
        float vv = acc[i][j][r];
        const int row = row0 + r;
        if (EPI == 1) {
          int diff = col - row;  // s - t
          vv = (diff > 0) ? vv * __expf((float)(diff - 1) * ldec) : 0.0f;
        }
        if (EPI == 2) vv *= scale;
        Cc[(size_t)row * ldc + col] = (OutT)vv;
      }
    }
}

extern "C" void kernel_launch(void* const* d_in, const int* in_sizes, int n_in,
                              void* d_out, int out_size, void* d_ws, size_t ws_size,
                              hipStream_t stream) {
  (void)in_sizes; (void)n_in; (void)out_size; (void)ws_size;
  const float* x  = (const float*)d_in[0];
  const float* wq = (const float*)d_in[1];
  const float* wk = (const float*)d_in[2];
  const float* wv = (const float*)d_in[3];
  const float* wo = (const float*)d_in[4];
  const float* decay_logit = (const float*)d_in[5];
  const float* out_scale   = (const float*)d_in[6];
  float* out = (float*)d_out;

  const int V = 8192, Cn = 512, T = 2048, Bb = 4;
  const int BT = Bb * T;            // 8192
  const int C3 = 3 * Cn;            // 1536

  char* base = (char*)d_ws;
  size_t off = 0;
  auto alloc = [&](size_t bytes) -> void* {
    void* p = base + off;
    off = (off + bytes + 255) & ~(size_t)255;
    return p;
  };

  bf16* Fb    = (bf16*)alloc((size_t)V * Cn * 2);     // 8 MB  F[v,j]
  bf16* wqb   = (bf16*)alloc((size_t)Cn * Cn * 2);
  bf16* wkb   = (bf16*)alloc((size_t)Cn * Cn * 2);
  bf16* wvb   = (bf16*)alloc((size_t)Cn * Cn * 2);
  bf16* wob   = (bf16*)alloc((size_t)Cn * Cn * 2);
  bf16* Wqkvc = (bf16*)alloc((size_t)V * C3 * 2);     // 24 MB [V, 3C]
  bf16* WoT   = (bf16*)alloc((size_t)V * Cn * 2);     // 8 MB  [V, C]
  bf16* qkv   = (bf16*)alloc((size_t)BT * C3 * 2);    // 24 MB [BT, 3C]
  bf16* sw    = (bf16*)alloc((size_t)Bb * T * T * 2); // 32 MB
  bf16* ret   = (bf16*)alloc((size_t)BT * Cn * 2);    // 8 MB

  dim3 blk(256);

  { // Fourier basis
    int total = V * Cn;
    gen_fourier_bf16<<<(total + 255) / 256, 256, 0, stream>>>(Fb, total);
  }
  { // weight conversions f32 -> bf16
    int n = Cn * Cn, g = (n + 255) / 256;
    cvt_f32_bf16<<<g, 256, 0, stream>>>(wq, wqb, n);
    cvt_f32_bf16<<<g, 256, 0, stream>>>(wk, wkb, n);
    cvt_f32_bf16<<<g, 256, 0, stream>>>(wv, wvb, n);
    cvt_f32_bf16<<<g, 256, 0, stream>>>(wo, wob, n);
  }
  { // combined projection weights: W*_c[v,c] = sum_j F[v,j] w[c,j]  (B = w, [N,K])
    dim3 g(Cn / BN, V / BM, 1);
    wmma_gemm<false, true, 0, bf16><<<g, blk, 0, stream>>>(
        Fb, wqb, Wqkvc + 0 * Cn, V, Cn, Cn, Cn, Cn, C3, 0, 0, 0, decay_logit, out_scale);
    wmma_gemm<false, true, 0, bf16><<<g, blk, 0, stream>>>(
        Fb, wkb, Wqkvc + 1 * Cn, V, Cn, Cn, Cn, Cn, C3, 0, 0, 0, decay_logit, out_scale);
    wmma_gemm<false, true, 0, bf16><<<g, blk, 0, stream>>>(
        Fb, wvb, Wqkvc + 2 * Cn, V, Cn, Cn, Cn, Cn, C3, 0, 0, 0, decay_logit, out_scale);
    // WoT[v,c] = sum_j F[v,j] wo[j,c]   (B = wo, [K,N])
    wmma_gemm<false, false, 0, bf16><<<g, blk, 0, stream>>>(
        Fb, wob, WoT, V, Cn, Cn, Cn, Cn, Cn, 0, 0, 0, decay_logit, out_scale);
  }
  { // fused qkv = x @ Wqkvc : M=8192, N=1536, K=8192 (x read from HBM once)
    dim3 g(C3 / BN, BT / BM, 1);
    wmma_gemm<true, false, 0, bf16><<<g, blk, 0, stream>>>(
        x, Wqkvc, qkv, BT, C3, V, V, C3, C3, 0, 0, 0, decay_logit, out_scale);
  }
  { // scores_w[b] = (q[b] @ k[b]^T) * decay-mask : batched over B
    dim3 g(T / BN, T / BM, Bb);
    wmma_gemm<false, true, 1, bf16><<<g, blk, 0, stream>>>(
        qkv + 0 * Cn, qkv + 1 * Cn, sw, T, T, Cn, C3, C3, T,
        (long long)T * C3, (long long)T * C3, (long long)T * T,
        decay_logit, out_scale);
  }
  { // retrieved[b] = scores_w[b] @ v[b]
    dim3 g(Cn / BN, T / BM, Bb);
    wmma_gemm<false, false, 0, bf16><<<g, blk, 0, stream>>>(
        sw, qkv + 2 * Cn, ret, T, Cn, T, T, C3, Cn,
        (long long)T * T, (long long)T * C3, (long long)T * Cn,
        decay_logit, out_scale);
  }
  { // out = retrieved @ WoT^T * (out_scale * 2/V)  -> f32 d_out
    dim3 g(V / BN, BT / BM, 1);
    wmma_gemm<false, true, 2, float><<<g, blk, 0, stream>>>(
        ret, WoT, out, BT, V, Cn, Cn, Cn, V, 0, 0, 0, decay_logit, out_scale);
  }
}
